// SortedDimWiseMLP_63136019251737
// MI455X (gfx1250) — matrix operand order repaired and checked by reading, not compiled, not run on hardware
//
#include <hip/hip_runtime.h>
#include <hip/hip_bf16.h>

typedef __bf16 bf16;
typedef __attribute__((ext_vector_type(16))) __bf16 v16bf;
typedef __attribute__((ext_vector_type(8)))  __bf16 v8bf;
typedef __attribute__((ext_vector_type(8)))  float  v8f;

#define M_ROWS 32768   // B*N
#define F_DIM  512
#define H_DIM  2048

// LDS tile K-stride in halfs (32 data + 8 pad -> 80B row pitch, 16B aligned,
// maps the 16 fragment rows onto distinct LDS banks)
#define TSTRIDE 40

// ---------- fragment loaders (bf16, 16x16x32, wave32) ----------
// A-matrix 16x32 (MxK): lanes 0-15 = rows M=0..15 with K {0..7, 16..23},
// lanes 16-31 same rows with K {8..15, 24..31}.
__device__ inline v16bf load_a_frag(const bf16* __restrict__ base, int ld,
                                    int row0, int k0, int lane) {
    const bf16* p = base + (size_t)(row0 + (lane & 15)) * ld + k0 + ((lane >> 4) << 3);
    v8bf lo = *(const v8bf*)(p);
    v8bf hi = *(const v8bf*)(p + 16);
    v16bf r;
#pragma unroll
    for (int i = 0; i < 8; ++i) { r[i] = lo[i]; r[i + 8] = hi[i]; }
    return r;
}

// B-matrix 32x16 (KxN), stored row-major [N][K]:
// lanes 0-15 = col n with K {0..7, 16..23}; lanes 16-31 = K {8..15, 24..31}.
__device__ inline v16bf load_b_frag(const bf16* __restrict__ base, int ld,
                                    int col0, int k0, int lane) {
    const bf16* p = base + (size_t)(col0 + (lane & 15)) * ld + k0 + ((lane >> 4) << 3);
    v8bf lo = *(const v8bf*)(p);
    v8bf hi = *(const v8bf*)(p + 16);
    v16bf r;
#pragma unroll
    for (int i = 0; i < 8; ++i) { r[i] = lo[i]; r[i + 8] = hi[i]; }
    return r;
}

// ---------- async global -> LDS tile copy (CDNA5 ASYNCcnt path) ----------
// Copies a ROWSx32-half K-slice into an LDS tile with TSTRIDE pitch.
// CHUNKS = ROWS*4 16-byte chunks, spread over 256 threads.
template <int CHUNKS>
__device__ inline void async_copy_tile(const bf16* __restrict__ gbase, int ld,
                                       int rc0, int k0, unsigned lds_base_bytes,
                                       int tid) {
#pragma unroll
    for (int it = 0; it < (CHUNKS + 255) / 256; ++it) {
        int c = tid + it * 256;            // chunk id
        if (c < CHUNKS) {
            int row = c >> 2;              // tile row
            int ko  = (c & 3) * 8;         // 0,8,16,24 (halfs)
            const bf16* g = gbase + (size_t)(rc0 + row) * ld + k0 + ko;
            unsigned l = lds_base_bytes + (unsigned)(row * (TSTRIDE * 2) + (c & 3) * 16);
            asm volatile("global_load_async_to_lds_b128 %0, %1, off"
                         :: "v"(l), "v"((unsigned long long)(size_t)g)
                         : "memory");
        }
    }
}

__device__ inline void wait_async_all() {
    asm volatile("s_wait_asynccnt 0x0" ::: "memory");
}

// ---------- kernel 0: fp32 -> bf16 convert ----------
__global__ __launch_bounds__(256) void cvt_bf16(const float* __restrict__ in,
                                                bf16* __restrict__ out, int n) {
    int i = blockIdx.x * 256 + threadIdx.x;
    if (i < n) out[i] = (bf16)in[i];
}

// ---------- kernel 1: per-row bitonic argsort + permute ----------
__global__ __launch_bounds__(256) void sort_rows(const float* __restrict__ x,
                                                 float* __restrict__ xs_f,
                                                 bf16* __restrict__ xs_b,
                                                 unsigned short* __restrict__ srt) {
    __shared__ float sv[F_DIM];
    __shared__ int   si[F_DIM];
    __shared__ float xo[F_DIM];
    __shared__ int   inv[F_DIM];

    const int tid = threadIdx.x;
    const size_t row = blockIdx.x;
    const float* xr = x + row * F_DIM;

    for (int i = tid; i < F_DIM; i += 256) {
        float v = xr[i];
        sv[i] = v; xo[i] = v; si[i] = i;
    }
    __syncthreads();

    for (int k = 2; k <= F_DIM; k <<= 1) {
        for (int j = k >> 1; j > 0; j >>= 1) {
            for (int i = tid; i < F_DIM; i += 256) {
                int ixj = i ^ j;
                if (ixj > i) {
                    bool up = ((i & k) == 0);
                    float a = sv[i], b = sv[ixj];
                    bool sw = up ? (a > b) : (a < b);
                    if (sw) {
                        sv[i] = b; sv[ixj] = a;
                        int t = si[i]; si[i] = si[ixj]; si[ixj] = t;
                    }
                }
            }
            __syncthreads();
        }
    }

    for (int k = tid; k < F_DIM; k += 256) inv[si[k]] = k;
    __syncthreads();

    for (int j = tid; j < F_DIM; j += 256) {
        float v = xo[inv[j]];
        size_t o = row * F_DIM + j;
        xs_f[o] = v;
        xs_b[o] = (bf16)v;
        srt[o]  = (unsigned short)si[j];
    }
}

// ---------- kernel 2: h = relu(xs * W1^T + b1) ----------
// 128x128 block tile; async double-buffered LDS staging; 8 waves of 32x64.
__global__ __launch_bounds__(256) void gemm1_relu(const bf16* __restrict__ xs,
                                                  const bf16* __restrict__ w1,
                                                  const float* __restrict__ b1,
                                                  bf16* __restrict__ h) {
    __shared__ __align__(16) bf16 smem[2][2][128 * TSTRIDE];   // 40 KB

    const int tid  = threadIdx.x;
    const int lane = tid & 31;
    const int wave = tid >> 5;
    const int rg = wave >> 1;      // 0..3
    const int cg = wave & 1;       // 0..1
    const int row0 = blockIdx.x * 128;
    const int col0 = blockIdx.y * 128;

    unsigned ldsA[2], ldsB[2];
#pragma unroll
    for (int b = 0; b < 2; ++b) {
        ldsA[b] = (unsigned)(size_t)(&smem[b][0][0]);
        ldsB[b] = (unsigned)(size_t)(&smem[b][1][0]);
    }

    v8f acc[2][4];
    const v8f vzero = {0.f, 0.f, 0.f, 0.f, 0.f, 0.f, 0.f, 0.f};
#pragma unroll
    for (int i = 0; i < 2; ++i)
#pragma unroll
        for (int j = 0; j < 4; ++j) acc[i][j] = vzero;

    async_copy_tile<512>(xs, F_DIM, row0, 0, ldsA[0], tid);
    async_copy_tile<512>(w1, F_DIM, col0, 0, ldsB[0], tid);

    const int KSTEPS = F_DIM / 32;    // 16
    for (int ks = 0; ks < KSTEPS; ++ks) {
        const int cur = ks & 1;
        wait_async_all();             // my stage-ks chunks landed in LDS
        __syncthreads();              // everyone's chunks landed; prev readers done

        if (ks + 1 < KSTEPS) {
            int k0 = (ks + 1) * 32;
            async_copy_tile<512>(xs, F_DIM, row0, k0, ldsA[1 - cur], tid);
            async_copy_tile<512>(w1, F_DIM, col0, k0, ldsB[1 - cur], tid);
        }

        const bf16* As = &smem[cur][0][0];
        const bf16* Bs = &smem[cur][1][0];

        v16bf a0 = load_a_frag(As, TSTRIDE, rg * 32,      0, lane);
        v16bf a1 = load_a_frag(As, TSTRIDE, rg * 32 + 16, 0, lane);
        v16bf bfrag[4];
#pragma unroll
        for (int j = 0; j < 4; ++j)
            bfrag[j] = load_b_frag(Bs, TSTRIDE, cg * 64 + j * 16, 0, lane);
#pragma unroll
        for (int j = 0; j < 4; ++j) {
            acc[0][j] = __builtin_amdgcn_wmma_f32_16x16x32_bf16(
                false, a0, false, bfrag[j], (short)0, acc[0][j], false, false);
            acc[1][j] = __builtin_amdgcn_wmma_f32_16x16x32_bf16(
                false, a1, false, bfrag[j], (short)0, acc[1][j], false, false);
        }
    }

    // epilogue: bias + relu -> bf16
    const int mlo = (lane >> 4) << 3;
    const int n   = lane & 15;
    const int wr0 = row0 + rg * 32;
    const int wc0 = col0 + cg * 64;
#pragma unroll
    for (int i = 0; i < 2; ++i)
#pragma unroll
        for (int j = 0; j < 4; ++j) {
            int c = wc0 + j * 16 + n;
            float bias = b1[c];
#pragma unroll
            for (int r = 0; r < 8; ++r) {
                int m = wr0 + i * 16 + r + mlo;
                float v = acc[i][j][r] + bias;
                h[(size_t)m * H_DIM + c] = (bf16)fmaxf(v, 0.f);
            }
        }
}

// ---------- kernel 3: y = h * W2^T + b2 + xs ; out = gather(y, srt) ----------
// 32-row x 512-col block tile; async double-buffered LDS staging;
// 8 waves each compute 32 rows x 64 cols (2x4 wmma accs).
__global__ __launch_bounds__(256) void gemm2_scatter(const bf16* __restrict__ h,
                                                     const bf16* __restrict__ w2,
                                                     const float* __restrict__ b2,
                                                     const float* __restrict__ xs_f,
                                                     const unsigned short* __restrict__ srt,
                                                     float* __restrict__ out) {
    __shared__ __align__(16) bf16 smA[2][32 * TSTRIDE];    //   5 KB
    __shared__ __align__(16) bf16 smB[2][512 * TSTRIDE];   //  80 KB
    __shared__ float yrow[32][F_DIM];                      //  64 KB

    const int tid  = threadIdx.x;
    const int lane = tid & 31;
    const int wave = tid >> 5;
    const int row0 = blockIdx.x * 32;
    const int col0 = wave * 64;

    unsigned ldsA[2], ldsB[2];
#pragma unroll
    for (int b = 0; b < 2; ++b) {
        ldsA[b] = (unsigned)(size_t)(&smA[b][0]);
        ldsB[b] = (unsigned)(size_t)(&smB[b][0]);
    }

    v8f acc[2][4];
    const v8f vzero = {0.f, 0.f, 0.f, 0.f, 0.f, 0.f, 0.f, 0.f};
#pragma unroll
    for (int i = 0; i < 2; ++i)
#pragma unroll
        for (int j = 0; j < 4; ++j) acc[i][j] = vzero;

    async_copy_tile<128>(h, H_DIM, row0, 0, ldsA[0], tid);   // A: 32 rows
    async_copy_tile<2048>(w2, H_DIM, 0,   0, ldsB[0], tid);  // B: all 512 cols

    const int KSTEPS = H_DIM / 32;    // 64
    for (int ks = 0; ks < KSTEPS; ++ks) {
        const int cur = ks & 1;
        wait_async_all();
        __syncthreads();

        if (ks + 1 < KSTEPS) {
            int k0 = (ks + 1) * 32;
            async_copy_tile<128>(h, H_DIM, row0, k0, ldsA[1 - cur], tid);
            async_copy_tile<2048>(w2, H_DIM, 0,   k0, ldsB[1 - cur], tid);
        }

        const bf16* As = &smA[cur][0];
        const bf16* Bs = &smB[cur][0];

        v16bf a0 = load_a_frag(As, TSTRIDE, 0,  0, lane);
        v16bf a1 = load_a_frag(As, TSTRIDE, 16, 0, lane);
        v16bf bfrag[4];
#pragma unroll
        for (int j = 0; j < 4; ++j)
            bfrag[j] = load_b_frag(Bs, TSTRIDE, col0 + j * 16, 0, lane);
#pragma unroll
        for (int j = 0; j < 4; ++j) {
            acc[0][j] = __builtin_amdgcn_wmma_f32_16x16x32_bf16(
                false, a0, false, bfrag[j], (short)0, acc[0][j], false, false);
            acc[1][j] = __builtin_amdgcn_wmma_f32_16x16x32_bf16(
                false, a1, false, bfrag[j], (short)0, acc[1][j], false, false);
        }
    }

    // epilogue: + b2 + fp32 skip into the LDS y-panel
    const int mlo = (lane >> 4) << 3;
    const int n   = lane & 15;
#pragma unroll
    for (int i = 0; i < 2; ++i)
#pragma unroll
        for (int j = 0; j < 4; ++j) {
            int c = col0 + j * 16 + n;
            float bias = b2[c];
#pragma unroll
            for (int r = 0; r < 8; ++r) {
                int m = i * 16 + r + mlo;
                float v = acc[i][j][r] + bias + xs_f[(size_t)(row0 + m) * F_DIM + c];
                yrow[m][c] = v;
            }
        }
    __syncthreads();

    // out[m][c] = y[m][srt[m][c]]
    for (int e = tid; e < 32 * F_DIM; e += 256) {
        int m = e >> 9;
        int c = e & (F_DIM - 1);
        size_t o = (size_t)(row0 + m) * F_DIM + c;
        out[o] = yrow[m][srt[o]];
    }
}

extern "C" void kernel_launch(void* const* d_in, const int* in_sizes, int n_in,
                              void* d_out, int out_size, void* d_ws, size_t ws_size,
                              hipStream_t stream) {
    const float* x  = (const float*)d_in[0];   // [32768, 512]
    const float* W1 = (const float*)d_in[1];   // [2048, 512]
    const float* b1 = (const float*)d_in[2];   // [2048]
    const float* W2 = (const float*)d_in[3];   // [512, 2048]
    const float* b2 = (const float*)d_in[4];   // [512]
    // d_in[5] = dim (==1, moveaxis is identity) -- ignored
    float* out = (float*)d_out;

    char* ws = (char*)d_ws;
    const size_t MB = 1024ull * 1024ull;
    bf16*           xs_b = (bf16*)(ws);                    //  32 MB
    float*          xs_f = (float*)(ws + 32 * MB);         //  64 MB
    unsigned short* srt  = (unsigned short*)(ws + 96 * MB);//  32 MB
    bf16*           hbuf = (bf16*)(ws + 128 * MB);         // 128 MB
    bf16*           w1b  = (bf16*)(ws + 256 * MB);         //   2 MB
    bf16*           w2b  = (bf16*)(ws + 258 * MB);         //   2 MB

    const int nw = H_DIM * F_DIM;
    cvt_bf16<<<(nw + 255) / 256, 256, 0, stream>>>(W1, w1b, nw);
    cvt_bf16<<<(nw + 255) / 256, 256, 0, stream>>>(W2, w2b, nw);

    sort_rows<<<M_ROWS, 256, 0, stream>>>(x, xs_f, xs_b, srt);

    gemm1_relu<<<dim3(M_ROWS / 128, H_DIM / 128), 256, 0, stream>>>(xs_b, w1b, b1, hbuf);

    gemm2_scatter<<<M_ROWS / 32, 256, 0, stream>>>(hbuf, w2b, b2, xs_f, srt, out);
}
